// LayerNormGRUCell_39745627357260
// MI455X (gfx1250) — matrix operand order repaired and checked by reading, not compiled, not run on hardware
//
#include <hip/hip_runtime.h>

#define BATCH 16384
#define KDIM  512           // K of each physical GEMM
#define HID   512
#define NCOL  2048          // [ pre_r(fused) | pre_z(fused) | x_n | h_n ]
#define KCAT  1024          // concatenated K ([x|h])
#define KT    64            // K per LDS stage (2 WMMA K-steps)
#define NKT   (KCAT / KT)   // 16 stages
#define TSTR  72            // LDS tile row stride in bf16 elements (144 B, 16B-aligned, bank-staggered)

typedef __attribute__((ext_vector_type(16))) __bf16 v16bf;
typedef __attribute__((ext_vector_type(8)))  float  v8f;
typedef int v4i __attribute__((vector_size(16)));   // matches async-LDS builtin param type

union FragU { uint4 q[2]; v16bf v; };

#if defined(__AMDGCN__) && __has_builtin(__builtin_amdgcn_global_load_async_to_lds_b128)
#define HAVE_ASYNC_LDS 1
#else
#define HAVE_ASYNC_LDS 0
#endif

// round-half-up float->bf16 pair pack via v_perm_b32 (3 VALU ops)
__device__ __forceinline__ unsigned pack_bf16x2(float a, float b) {
  unsigned ua = __float_as_uint(a) + 0x8000u;
  unsigned ub = __float_as_uint(b) + 0x8000u;
  return __builtin_amdgcn_perm(ub, ua, 0x07060302u);  // { ub[31:16], ua[31:16] }
}

__device__ __forceinline__ unsigned short f2bf(float a) {
  return (unsigned short)((__float_as_uint(a) + 0x8000u) >> 16);
}

__device__ __forceinline__ float bf2f(unsigned short u) {
  union { unsigned u; float f; } c;
  c.u = ((unsigned)u) << 16;
  return c.f;
}

// 16-byte global -> LDS copy; async (ASYNCcnt) when the toolchain exposes it
__device__ __forceinline__ void cp16_g2l(const unsigned short* gsrc, unsigned short* ldst) {
#if HAVE_ASYNC_LDS
  __builtin_amdgcn_global_load_async_to_lds_b128(
      (__attribute__((address_space(1))) v4i*)(unsigned short*)gsrc,
      (__attribute__((address_space(3))) v4i*)ldst, 0, 0);
#else
  *(uint4*)ldst = *(const uint4*)gsrc;
#endif
}

#if HAVE_ASYNC_LDS
__device__ __forceinline__ void wait_async0() {
#if __has_builtin(__builtin_amdgcn_s_wait_asynccnt)
  __builtin_amdgcn_s_wait_asynccnt(0);
#else
  asm volatile("s_wait_asynccnt 0" ::: "memory");
#endif
}
#endif

// A fragment (16x32 bf16): lanes 0-15 -> M; K chunks {kb..kb+7, kb+16..kb+23}, kb=(lane>>4)*8
__device__ __forceinline__ v16bf frag_a(const unsigned short* lds, int r0, int kc, int lane) {
  const unsigned short* p = lds + (r0 + (lane & 15)) * TSTR + kc * 32 + ((lane >> 4) << 3);
  FragU u;
  u.q[0] = *(const uint4*)(p);
  u.q[1] = *(const uint4*)(p + 16);
  return u.v;
}

// B fragment (32x16 bf16, LDS row = N): K chunks {kb..kb+7, kb+16..kb+23}, kb=(lane>>4)*8
__device__ __forceinline__ v16bf frag_b(const unsigned short* lds, int r0, int kc, int lane) {
  const unsigned short* p = lds + (r0 + (lane & 15)) * TSTR + kc * 32 + ((lane >> 4) << 3);
  FragU u;
  u.q[0] = *(const uint4*)(p);
  u.q[1] = *(const uint4*)(p + 16);
  return u.v;
}

// ---- pre-pass: Xb[b, 0:512)=bf16(x[b,:]), Xb[b, 512:1024)=bf16(h[b,:])
__global__ __launch_bounds__(256)
void pack_x_kernel(const float* __restrict__ x, const float* __restrict__ h,
                   unsigned short* __restrict__ Xb) {
  const int idx = blockIdx.x * 256 + threadIdx.x;   // one float4 chunk
  const int b   = idx >> 8;                         // 256 chunks per row of 1024
  const int c4  = (idx & 255) * 4;
  const float* src = (c4 < KDIM) ? (x + (size_t)b * KDIM + c4)
                                 : (h + (size_t)b * KDIM + (c4 - KDIM));
  float4 v = *(const float4*)src;
  uint2 p;
  p.x = pack_bf16x2(v.x, v.y);
  p.y = pack_bf16x2(v.z, v.w);
  *(uint2*)(Xb + (size_t)b * KCAT + c4) = p;
}

// ---- pre-pass: materialize virtual big-W (2048 x 1024, bf16, zero-padded)
//  rows 0:1024   : [ Wi_row | Wh_row ]     (r, z fused)
//  rows 1024:1536: [ Wi_row |   0    ]     (x_n)
//  rows 1536:2048: [   0    | Wh_row-512 ] (h_n)
__global__ __launch_bounds__(256)
void pack_w_kernel(const float* __restrict__ Wi, const float* __restrict__ Wh,
                   unsigned short* __restrict__ Wb) {
  const int idx = blockIdx.x * 256 + threadIdx.x;
  const int j   = idx >> 8;
  const int c4  = (idx & 255) * 4;
  const int seg = j >> 9;
  float4 v = {0.f, 0.f, 0.f, 0.f};
  if (c4 < KDIM) {
    if (seg < 3) v = *(const float4*)(Wi + (size_t)j * KDIM + c4);
  } else {
    const int cc = c4 - KDIM;
    if (seg < 2)       v = *(const float4*)(Wh + (size_t)j * KDIM + cc);
    else if (seg == 3) v = *(const float4*)(Wh + (size_t)(j - 512) * KDIM + cc);
  }
  uint2 p;
  p.x = pack_bf16x2(v.x, v.y);
  p.y = pack_bf16x2(v.z, v.w);
  *(uint2*)(Wb + (size_t)j * KCAT + c4) = p;
}

// ---- pure bf16 GEMM: G[16384, 2048] = Xb[16384,1024] @ Wb[2048,1024]^T
__global__ __launch_bounds__(256)
void gru_gemm_kernel(const unsigned short* __restrict__ Xb,
                     const unsigned short* __restrict__ Wb,
                     unsigned short* __restrict__ G) {
  __shared__ unsigned short ldsA[2][128 * TSTR];
  __shared__ unsigned short ldsB[2][128 * TSTR];

  const int tid  = threadIdx.x;
  const int lane = tid & 31;
  const int w    = tid >> 5;
  const int wm   = w & 3;    // 4 waves along M (32 rows each)
  const int wn   = w >> 2;   // 2 waves along N (64 cols each)

  const int m0 = blockIdx.x * 128;
  const int n0 = blockIdx.y * 128;

  // staging: tile is 128 rows x 64 bf16 = 1024 16B-chunks; 4 per thread
  const int q8   = (tid & 7) * 8;       // bf16 offset of this thread's 16B chunk in a row
  const int row0 = tid >> 3;            // base row (+ i*32)

  auto stage = [&](int buf, int k0) {
#pragma unroll
    for (int i = 0; i < 4; ++i) {
      const int row = i * 32 + row0;
      cp16_g2l(Xb + (size_t)(m0 + row) * KCAT + k0 + q8, &ldsA[buf][row * TSTR + q8]);
      cp16_g2l(Wb + (size_t)(n0 + row) * KCAT + k0 + q8, &ldsB[buf][row * TSTR + q8]);
    }
  };

  v8f c[2][4];
#pragma unroll
  for (int i = 0; i < 2; ++i)
#pragma unroll
    for (int j = 0; j < 4; ++j)
      c[i][j] = (v8f){0.f, 0.f, 0.f, 0.f, 0.f, 0.f, 0.f, 0.f};

  // prologue
  stage(0, 0);
#if HAVE_ASYNC_LDS
  wait_async0();
#endif
  __syncthreads();

  for (int kt = 0; kt < NKT; ++kt) {
    const int cur = kt & 1;
    const bool have_next = (kt + 1 < NKT);

    if (have_next) stage(cur ^ 1, (kt + 1) * KT);   // overlaps with WMMAs below

#pragma unroll
    for (int kc = 0; kc < 2; ++kc) {
      v16bf a[2], b[4];
#pragma unroll
      for (int tm = 0; tm < 2; ++tm) a[tm] = frag_a(ldsA[cur], wm * 32 + tm * 16, kc, lane);
#pragma unroll
      for (int tn = 0; tn < 4; ++tn) b[tn] = frag_b(ldsB[cur], wn * 64 + tn * 16, kc, lane);
#pragma unroll
      for (int tm = 0; tm < 2; ++tm)
#pragma unroll
        for (int tn = 0; tn < 4; ++tn)
          c[tm][tn] = __builtin_amdgcn_wmma_f32_16x16x32_bf16(
              false, a[tm], false, b[tn], (short)0, c[tm][tn], false, false);
    }

#if HAVE_ASYNC_LDS
    if (have_next) wait_async0();
#endif
    __syncthreads();
  }

  // epilogue: bf16 stores. C/D layout: VGPR i -> M=(lane>=16?8:0)+i, lane&15 -> N
#pragma unroll
  for (int tm = 0; tm < 2; ++tm) {
    const int rbase = m0 + wm * 32 + tm * 16 + (lane >> 4) * 8;
#pragma unroll
    for (int tn = 0; tn < 4; ++tn) {
      const int col = n0 + wn * 64 + tn * 16 + (lane & 15);
#pragma unroll
      for (int i = 0; i < 8; ++i)
        G[(size_t)(rbase + i) * NCOL + col] = f2bf(c[tm][tn][i]);
    }
  }
}

__device__ __forceinline__ float2 block_reduce2(float s, float q, float* red) {
#pragma unroll
  for (int off = 16; off > 0; off >>= 1) {
    s += __shfl_xor(s, off, 32);
    q += __shfl_xor(q, off, 32);
  }
  const int w = threadIdx.x >> 5;
  if ((threadIdx.x & 31) == 0) { red[w * 2] = s; red[w * 2 + 1] = q; }
  __syncthreads();
  float ts = 0.f, tq = 0.f;
#pragma unroll
  for (int i = 0; i < 8; ++i) { ts += red[i * 2]; tq += red[i * 2 + 1]; }
  __syncthreads();
  return make_float2(ts, tq);
}

__device__ __forceinline__ float sigmoidf_(float v) { return 1.0f / (1.0f + __expf(-v)); }

__global__ __launch_bounds__(256)
void gru_gate_kernel(const unsigned short* __restrict__ G, const float* __restrict__ h,
                     const float* __restrict__ g_r, const float* __restrict__ b_r,
                     const float* __restrict__ g_z, const float* __restrict__ b_z,
                     const float* __restrict__ g_n, const float* __restrict__ b_n,
                     float* __restrict__ out) {
  __shared__ float red[16];
  const int b   = blockIdx.x;
  const int tid = threadIdx.x;
  const int j0 = tid, j1 = tid + 256;
  const unsigned short* row = G + (size_t)b * NCOL;
  const float invH = 1.0f / (float)HID;
  const float eps  = 1e-5f;

  // r gate
  float p0 = bf2f(row[j0]), p1 = bf2f(row[j1]);
  float2 sr = block_reduce2(p0 + p1, p0 * p0 + p1 * p1, red);
  float mean = sr.x * invH;
  float inv  = rsqrtf(sr.y * invH - mean * mean + eps);
  float r0 = sigmoidf_((p0 - mean) * inv * g_r[j0] + b_r[j0]);
  float r1 = sigmoidf_((p1 - mean) * inv * g_r[j1] + b_r[j1]);

  // z gate
  p0 = bf2f(row[HID + j0]); p1 = bf2f(row[HID + j1]);
  sr = block_reduce2(p0 + p1, p0 * p0 + p1 * p1, red);
  mean = sr.x * invH;
  inv  = rsqrtf(sr.y * invH - mean * mean + eps);
  float z0 = sigmoidf_((p0 - mean) * inv * g_z[j0] + b_z[j0]);
  float z1 = sigmoidf_((p1 - mean) * inv * g_z[j1] + b_z[j1]);

  // n gate: x_n + r * h_n
  p0 = bf2f(row[2 * HID + j0]) + r0 * bf2f(row[3 * HID + j0]);
  p1 = bf2f(row[2 * HID + j1]) + r1 * bf2f(row[3 * HID + j1]);
  sr = block_reduce2(p0 + p1, p0 * p0 + p1 * p1, red);
  mean = sr.x * invH;
  inv  = rsqrtf(sr.y * invH - mean * mean + eps);
  float n0 = tanhf((p0 - mean) * inv * g_n[j0] + b_n[j0]);
  float n1 = tanhf((p1 - mean) * inv * g_n[j1] + b_n[j1]);

  const float h0 = h[(size_t)b * HID + j0];
  const float h1 = h[(size_t)b * HID + j1];
  out[(size_t)b * HID + j0] = (1.0f - z0) * n0 + z0 * h0;
  out[(size_t)b * HID + j1] = (1.0f - z1) * n1 + z1 * h1;
}

extern "C" void kernel_launch(void* const* d_in, const int* in_sizes, int n_in,
                              void* d_out, int out_size, void* d_ws, size_t ws_size,
                              hipStream_t stream) {
  const float* x   = (const float*)d_in[0];
  const float* h   = (const float*)d_in[1];
  const float* Wi  = (const float*)d_in[2];
  const float* Wh  = (const float*)d_in[3];
  const float* g_r = (const float*)d_in[4];
  const float* b_r = (const float*)d_in[5];
  const float* g_z = (const float*)d_in[6];
  const float* b_z = (const float*)d_in[7];
  const float* g_n = (const float*)d_in[8];
  const float* b_n = (const float*)d_in[9];
  float* out = (float*)d_out;

  // workspace layout (100 MiB total):
  unsigned short* G  = (unsigned short*)d_ws;                       // 64 MiB gates
  unsigned short* Xb = G + (size_t)BATCH * NCOL;                    // 32 MiB bf16 [x|h]
  unsigned short* Wb = Xb + (size_t)BATCH * KCAT;                   //  4 MiB bf16 big-W

  pack_x_kernel<<<(BATCH * KCAT / 4) / 256, 256, 0, stream>>>(x, h, Xb);
  pack_w_kernel<<<(NCOL * KCAT / 4) / 256, 256, 0, stream>>>(Wi, Wh, Wb);

  dim3 grid1(BATCH / 128, NCOL / 128);
  gru_gemm_kernel<<<grid1, 256, 0, stream>>>(Xb, Wb, G);
  gru_gate_kernel<<<BATCH, 256, 0, stream>>>(G, h, g_r, b_r, g_z, b_z, g_n, b_n, out);
}
